// ChebyshevKANLayer_72868415144092
// MI455X (gfx1250) — compile-verified, matrix operand order checked by reading, not run on hardware
//
#include <hip/hip_runtime.h>
#include <cstdint>

// Problem constants (from reference)
#define B_ROWS 4096
#define IN_F   1024
#define OUT_F  1024
#define DEG    8
#define KDIM   (IN_F * (DEG + 1))   // 9216

typedef __attribute__((ext_vector_type(16))) _Float16 v16h;
typedef __attribute__((ext_vector_type(8)))  float    v8f;
// GCC-style vector exactly matching the async builtin's parameter pointee type
typedef int gv4i __attribute__((__vector_size__(4 * sizeof(int))));

// ---- gfx1250 async global->LDS path (guarded; fallback = load + ds_store) ----
#if defined(__has_builtin)
#if __has_builtin(__builtin_amdgcn_global_load_async_to_lds_b128) && \
    __has_builtin(__builtin_amdgcn_s_wait_asynccnt)
#define HAVE_ASYNC_LDS 1
#endif
#endif

__device__ __forceinline__ void copy16_g2l(const void* g, void* l) {
#if defined(HAVE_ASYNC_LDS)
  __builtin_amdgcn_global_load_async_to_lds_b128(
      (__attribute__((address_space(1))) gv4i*)(uintptr_t)g,
      (__attribute__((address_space(3))) gv4i*)(uintptr_t)l,
      0, 0);
#else
  *(uint4*)l = *(const uint4*)g;
#endif
}

__device__ __forceinline__ void wait_async_copies() {
#if defined(HAVE_ASYNC_LDS)
  __builtin_amdgcn_s_wait_asynccnt(0);
#endif
}

// ---------------------------------------------------------------------------
// Kernel 1: per-row min/max + build augmented activation matrix A [B, KDIM] f16
//   A[b, 0*1024 + i]      = silu(x[b,i])
//   A[b, (d+1)*1024 + i]  = x[b,i] * T_d(xs[b,i]),  d = 0..7
// ---------------------------------------------------------------------------
__global__ __launch_bounds__(256)
void kan_prepA(const float* __restrict__ x, _Float16* __restrict__ A) {
  __shared__ float smn[256];
  __shared__ float smx[256];
  const int b   = blockIdx.x;
  const int tid = threadIdx.x;
  const float* xr = x + (size_t)b * IN_F;

  float xv[4];
  float mn = 3.402823466e38f, mx = -3.402823466e38f;
#pragma unroll
  for (int j = 0; j < 4; ++j) {
    xv[j] = xr[tid + j * 256];
    mn = fminf(mn, xv[j]);
    mx = fmaxf(mx, xv[j]);
  }
  smn[tid] = mn;
  smx[tid] = mx;
  __syncthreads();
#pragma unroll
  for (int s = 128; s > 0; s >>= 1) {
    if (tid < s) {
      smn[tid] = fminf(smn[tid], smn[tid + s]);
      smx[tid] = fmaxf(smx[tid], smx[tid + s]);
    }
    __syncthreads();
  }
  const float rmn = smn[0];
  const float rmx = smx[0];
  const float inv = 2.0f / (rmx - rmn);

  _Float16* Ar = A + (size_t)b * KDIM;
#pragma unroll
  for (int j = 0; j < 4; ++j) {
    const int i  = tid + j * 256;
    const float v = xv[j];
    // silu path
    Ar[i] = (_Float16)(v / (1.0f + __expf(-v)));
    // chebyshev path
    const float xs = (v - rmn) * inv - 1.0f;
    float tm2 = 1.0f;    // T0
    float tm1 = xs;      // T1
    Ar[IN_F + i]     = (_Float16)(v * tm2);
    Ar[2 * IN_F + i] = (_Float16)(v * tm1);
#pragma unroll
    for (int d = 2; d < DEG; ++d) {
      const float t = 2.0f * xs * tm1 - tm2;
      tm2 = tm1;
      tm1 = t;
      Ar[(d + 1) * IN_F + i] = (_Float16)(v * t);
    }
  }
}

// ---------------------------------------------------------------------------
// Kernel 2: pack weights, transposed: Wt[o, k] f16, k = dblk*1024 + i
// ---------------------------------------------------------------------------
__global__ __launch_bounds__(256)
void kan_prepW(const float* __restrict__ Wb, const float* __restrict__ Wc,
               _Float16* __restrict__ Wt) {
  const size_t t = (size_t)blockIdx.x * 256 + threadIdx.x;
  const int k = (int)(t % KDIM);
  const int o = (int)(t / KDIM);
  const int dblk = k >> 10;      // k / 1024
  const int i    = k & 1023;     // k % 1024
  float w;
  if (dblk == 0) {
    w = Wb[(size_t)i * OUT_F + o];
  } else {
    w = Wc[((size_t)i * OUT_F + o) * DEG + (dblk - 1)];
  }
  Wt[t] = (_Float16)w;
}

// ---------------------------------------------------------------------------
// Kernel 3: C[4096,1024] = A[4096,9216] @ Wt^T   (Wt stored [1024, 9216])
// Block: 128 threads = 4 waves (2x2). Block tile 128x128, BK=32.
// Wave tile 64x64 = 4x4 wmma tiles -> 16 wmma : 16 ds_load_b128 per K-step.
// Double-buffered LDS: async copies for tile i+1 overlap WMMA of tile i.
// ---------------------------------------------------------------------------
__global__ __launch_bounds__(128)
void kan_gemm(const _Float16* __restrict__ A, const _Float16* __restrict__ Wt,
              float* __restrict__ C) {
  constexpr int BM  = 128;
  constexpr int BN  = 128;
  constexpr int BK  = 32;
  constexpr int BKP = 40;   // padded LDS row stride (f16) -> conflict-free ds_load_b128

  __shared__ __align__(16) _Float16 As[2][BM][BKP];
  __shared__ __align__(16) _Float16 Bs[2][BN][BKP];

  const int tid  = threadIdx.x;
  const int lane = tid & 31;
  const int wid  = tid >> 5;      // 0..3
  const int wm   = wid >> 1;      // 0..1  (M direction)
  const int wn   = wid & 1;       // 0..1  (N direction)
  const int lh   = lane & 15;     // lane within half
  const int sel  = lane >> 4;     // which K-half group this lane owns

  const int mBlock = blockIdx.y * BM;
  const int nBlock = blockIdx.x * BN;

  v8f acc[4][4] = {};

  union Frag {
    uint4 u[2];
    v16h  v;
  };

  // ---- prologue: issue async copies for first K tile into buffer 0
#pragma unroll
  for (int j = 0; j < 4; ++j) {
    const int chunk = tid * 4 + j;        // 0..511
    const int row   = chunk >> 2;         // 0..127
    const int col   = (chunk & 3) * 8;    // 0,8,16,24 (f16 units)
    copy16_g2l(&A[(size_t)(mBlock + row) * KDIM + col], &As[0][row][col]);
    copy16_g2l(&Wt[(size_t)(nBlock + row) * KDIM + col], &Bs[0][row][col]);
  }

  int buf = 0;
  for (int k0 = 0; k0 < KDIM; k0 += BK) {
    // wait for this buffer's copies (per-wave), then make them block-visible
    wait_async_copies();
    __syncthreads();

    // ---- issue async copies for the NEXT K tile into the other buffer;
    //      they stay in flight underneath the 16 WMMAs below.
    const int kn = k0 + BK;
    if (kn < KDIM) {
#pragma unroll
      for (int j = 0; j < 4; ++j) {
        const int chunk = tid * 4 + j;
        const int row   = chunk >> 2;
        const int col   = (chunk & 3) * 8;
        copy16_g2l(&A[(size_t)(mBlock + row) * KDIM + kn + col],
                   &As[buf ^ 1][row][col]);
        copy16_g2l(&Wt[(size_t)(nBlock + row) * KDIM + kn + col],
                   &Bs[buf ^ 1][row][col]);
      }
      // pull the K tile after that toward L2/LDS-adjacent caches
      if (kn + BK < KDIM) {
        __builtin_prefetch(&A[(size_t)(mBlock + tid) * KDIM + kn + BK], 0, 3);
        __builtin_prefetch(&Wt[(size_t)(nBlock + tid) * KDIM + kn + BK], 0, 3);
      }
    }

    // ---- fragments (16-bit A/B layout: lanes0-15 K{0..7,16..23},
    //      lanes16-31 K{8..15,24..31}); 2x ds_load_b128 each
    Frag bf[4];
#pragma unroll
    for (int nt = 0; nt < 4; ++nt) {
      const int col = wn * 64 + nt * 16 + lh;
      bf[nt].u[0] = *(const uint4*)&Bs[buf][col][sel * 8];
      bf[nt].u[1] = *(const uint4*)&Bs[buf][col][sel * 8 + 16];
    }
#pragma unroll
    for (int mt = 0; mt < 4; ++mt) {
      Frag af;
      const int row = wm * 64 + mt * 16 + lh;
      af.u[0] = *(const uint4*)&As[buf][row][sel * 8];
      af.u[1] = *(const uint4*)&As[buf][row][sel * 8 + 16];
#pragma unroll
      for (int nt = 0; nt < 4; ++nt) {
        acc[mt][nt] = __builtin_amdgcn_wmma_f32_16x16x32_f16(
            false, af.v, false, bf[nt].v,
            (short)0, acc[mt][nt], false, false);
      }
    }
    buf ^= 1;
  }

  // ---- epilogue: 16x16 f32 C layout: VGPR r -> (M = r + 8*sel), N = lh
#pragma unroll
  for (int mt = 0; mt < 4; ++mt) {
#pragma unroll
    for (int nt = 0; nt < 4; ++nt) {
      const int col   = nBlock + wn * 64 + nt * 16 + lh;
      const int rbase = mBlock + wm * 64 + mt * 16 + sel * 8;
#pragma unroll
      for (int r = 0; r < 8; ++r) {
        C[(size_t)(rbase + r) * OUT_F + col] = acc[mt][nt][r];
      }
    }
  }
}

// ---------------------------------------------------------------------------
extern "C" void kernel_launch(void* const* d_in, const int* in_sizes, int n_in,
                              void* d_out, int out_size, void* d_ws, size_t ws_size,
                              hipStream_t stream) {
  const float* x  = (const float*)d_in[0];  // [4096, 1024]
  const float* wb = (const float*)d_in[1];  // [1024, 1024]
  const float* wc = (const float*)d_in[2];  // [1024, 1024, 8]
  float* out = (float*)d_out;               // [4096, 1024]

  _Float16* A  = (_Float16*)d_ws;                                   // 75.5 MB
  _Float16* Wt = (_Float16*)((char*)d_ws +
                             (size_t)B_ROWS * KDIM * sizeof(_Float16)); // +18.9 MB

  kan_prepA<<<B_ROWS, 256, 0, stream>>>(x, A);
  kan_prepW<<<(OUT_F * KDIM) / 256, 256, 0, stream>>>(wb, wc, Wt);

  dim3 grid(OUT_F / 128, B_ROWS / 128);  // 8 x 32 = 256 blocks, 4 waves each
  kan_gemm<<<grid, 128, 0, stream>>>(A, Wt, out);
}
